// GaussianVoxelizer_50225347559539
// MI455X (gfx1250) — matrix operand order repaired and checked by reading, not compiled, not run on hardware
//
#include <hip/hip_runtime.h>
#include <hip/hip_bf16.h>

typedef __attribute__((ext_vector_type(16))) _Float16 v16h;
typedef __attribute__((ext_vector_type(8)))  float    v8f;
typedef __attribute__((ext_vector_type(4)))  float    v4f;

#define NG    8192
#define NF    32
#define HGRID 160
#define WGRID 160
#define DGRID 16
#define NVOX  (HGRID * WGRID * DGRID)
#define PSTR  16          // floats per packed gaussian record
#define NTILE 3200        // = gridDim of main kernel; tile == blockIdx.x
#define CAP   320         // per-tile gaussian list capacity (expected ~83)

// ---------------------------------------------------------------------------
// Kernel 1: per-gaussian preprocessing.
// Packs: [mu0 mu1 mu2 lopac | A00 A11 A22 B01 | B02 B12 s3x s3y | s3z 0 0 0]
// A/B are C^-1 entries pre-scaled by 0.5*log2(e) (off-diag includes the x2),
// lopac = log2(opacity), s3 = 3*sigma.
// ---------------------------------------------------------------------------
__global__ __launch_bounds__(256) void gv_prep(const float* __restrict__ means,
                                               const float* __restrict__ opac,
                                               const float* __restrict__ scales,
                                               const float* __restrict__ rot,
                                               float* __restrict__ gp)
{
    int g = blockIdx.x * 256 + threadIdx.x;
    if (g >= NG) return;

    float qw = rot[g * 4 + 0], qx = rot[g * 4 + 1];
    float qy = rot[g * 4 + 2], qz = rot[g * 4 + 3];
    float rin = rsqrtf(qw * qw + qx * qx + qy * qy + qz * qz);
    qw *= rin; qx *= rin; qy *= rin; qz *= rin;

    float R00 = 1.0f - 2.0f * (qy * qy + qz * qz);
    float R01 = 2.0f * (qx * qy - qw * qz);
    float R02 = 2.0f * (qx * qz + qw * qy);
    float R10 = 2.0f * (qx * qy + qw * qz);
    float R11 = 1.0f - 2.0f * (qx * qx + qz * qz);
    float R12 = 2.0f * (qy * qz - qw * qx);
    float R20 = 2.0f * (qx * qz - qw * qy);
    float R21 = 2.0f * (qy * qz + qw * qx);
    float R22 = 1.0f - 2.0f * (qx * qx + qy * qy);

    float sx = scales[g * 3 + 0], sy = scales[g * 3 + 1], sz = scales[g * 3 + 2];
    float s0 = sx * sx, s1 = sy * sy, s2 = sz * sz;

    float c00 = R00 * R00 * s0 + R01 * R01 * s1 + R02 * R02 * s2;
    float c01 = R00 * R10 * s0 + R01 * R11 * s1 + R02 * R12 * s2;
    float c02 = R00 * R20 * s0 + R01 * R21 * s1 + R02 * R22 * s2;
    float c11 = R10 * R10 * s0 + R11 * R11 * s1 + R12 * R12 * s2;
    float c12 = R10 * R20 * s0 + R11 * R21 * s1 + R12 * R22 * s2;
    float c22 = R20 * R20 * s0 + R21 * R21 * s1 + R22 * R22 * s2;

    float M00 = c11 * c22 - c12 * c12;
    float M01 = c02 * c12 - c01 * c22;
    float M02 = c01 * c12 - c02 * c11;
    float det = c00 * M00 + c01 * M01 + c02 * M02;
    float id  = 1.0f / det;
    float i00 = M00 * id;
    float i01 = M01 * id;
    float i02 = M02 * id;
    float i11 = (c00 * c22 - c02 * c02) * id;
    float i12 = (c01 * c02 - c00 * c12) * id;
    float i22 = (c00 * c11 - c01 * c01) * id;

    const float HL2E = 0.72134752044448170f;   // 0.5 * log2(e)
    float* p = gp + (size_t)g * PSTR;
    p[0]  = means[g * 3 + 0];
    p[1]  = means[g * 3 + 1];
    p[2]  = means[g * 3 + 2];
    p[3]  = __builtin_log2f(opac[g]);
    p[4]  = i00 * HL2E;
    p[5]  = i11 * HL2E;
    p[6]  = i22 * HL2E;
    p[7]  = i01 * (2.0f * HL2E);
    p[8]  = i02 * (2.0f * HL2E);
    p[9]  = i12 * (2.0f * HL2E);
    p[10] = 3.0f * sqrtf(c00);
    p[11] = 3.0f * sqrtf(c11);
    p[12] = 3.0f * sqrtf(c22);
    p[13] = 0.0f; p[14] = 0.0f; p[15] = 0.0f;
}

// ---------------------------------------------------------------------------
// Kernel 2: feature transpose + f16 convert:  Ft[n][g] = (f16) F[g][n]
// ---------------------------------------------------------------------------
__global__ __launch_bounds__(256) void gv_featT(const float* __restrict__ feat,
                                                _Float16* __restrict__ ft)
{
    int t = blockIdx.x * 256 + threadIdx.x;   // t = n*NG + g
    int g = t & (NG - 1);
    int n = t >> 13;                          // NG == 2^13
    ft[t] = (_Float16)feat[g * NF + n];
}

// ---------------------------------------------------------------------------
// Binning: tile = blockIdx.x of the main kernel = i*20 + b, where the tile
// covers x-column i (single x voxel center) and j in [8b, 8b+7], full z.
// ---------------------------------------------------------------------------
__global__ __launch_bounds__(256) void gv_zero(int* __restrict__ counts)
{
    int t = blockIdx.x * 256 + threadIdx.x;
    if (t < NTILE) counts[t] = 0;
}

__global__ __launch_bounds__(256) void gv_count(const float* __restrict__ gp,
                                                int* __restrict__ counts,
                                                int* __restrict__ list)
{
    int g = blockIdx.x * 256 + threadIdx.x;
    if (g >= NG) return;
    const float* p = gp + (size_t)g * PSTR;
    float mu0 = p[0], mu1 = p[1], mu2 = p[2];
    float s3x = p[10], s3y = p[11], s3z = p[12];

    // z reject: voxel z-centers span [-0.8, 5.2]
    if (mu2 - s3z > 5.2f + 1e-4f || mu2 + s3z < -0.8f - 1e-4f) return;

    // x: center xc(i) = 0.4*i - 31.8 ; |xc - mu0| <= s3x
    int i0 = (int)ceilf ((mu0 - s3x + 31.8f) * 2.5f - 1e-4f);
    int i1 = (int)floorf((mu0 + s3x + 31.8f) * 2.5f + 1e-4f);
    i0 = i0 < 0 ? 0 : i0;  i1 = i1 > HGRID - 1 ? HGRID - 1 : i1;
    // y: block b covers yc in [3.2b - 31.8, 3.2b - 29.0]
    int b0 = (int)ceilf ((mu1 - s3y + 29.0f) * 0.3125f - 1e-4f);
    int b1 = (int)floorf((mu1 + s3y + 31.8f) * 0.3125f + 1e-4f);
    b0 = b0 < 0 ? 0 : b0;  b1 = b1 > 19 ? 19 : b1;

    for (int i = i0; i <= i1; ++i)
        for (int b = b0; b <= b1; ++b) {
            int tile = i * 20 + b;
            int slot = atomicAdd(&counts[tile], 1);
            if (slot < CAP) list[tile * CAP + slot] = g;
        }
}

// ---------------------------------------------------------------------------
// Kernel 3 (binned): 256 threads = 8 waves; each wave owns one z-column of 16
// voxels (same x,y across lanes -- only z varies). Per 32-gaussian list chunk:
//  stage:   32 gaussian ids -> LDS; gather their 32x32 f16 features -> LDS B.
//  phase 1: each lane reduces ONE listed gaussian to z-independent coeffs
//           {mu2, A22, c1, e0, s3z}; x/y mask and list-padding folded into
//           e0 = -1e30 (exp2 -> 0, so padded B content is harmless).
//  phase 2: 16 A-elements: broadcast LDS read + sub + 2 FMA + exp2 + cmp +
//           cndmask, A in contiguous k order (k = 16*laneHi + e, as B); B read from LDS.
//  Then two v_wmma_f32_16x16x32_f16 accumulate the 16x16 f32 tiles.
// ---------------------------------------------------------------------------
__global__ __launch_bounds__(256) void gv_main_binned(const float* __restrict__ gp,
                                                      const _Float16* __restrict__ ft,
                                                      const int* __restrict__ counts,
                                                      const int* __restrict__ list,
                                                      float* __restrict__ out)
{
    __shared__ float    smem[8 * 32 * 8];      // per-wave coefficient slots (8KB)
    __shared__ _Float16 Bls[32 * 32];          // [n][kk] staged features (2KB)
    __shared__ int      idxLds[32];            // chunk gaussian ids (-1 = pad)

    const int tid    = threadIdx.x;
    const int lane   = tid & 31;
    const int wave   = tid >> 5;
    const int laneHi = lane >> 4;
    const int m      = lane & 15;
    const int vBase  = blockIdx.x * 128 + wave * 16;

    // voxel coords: tile shares i; j = j0 + wave_offset... derive from v
    int v   = vBase + m;
    int i   = v / (WGRID * DGRID);
    int rem = v - i * (WGRID * DGRID);
    int j   = rem >> 4;
    int k   = rem & 15;
    float px = ((float)i + 0.5f) * 0.4f - 32.0f;
    float py = ((float)j + 0.5f) * 0.4f - 32.0f;
    float pz = ((float)k + 0.5f) * 0.4f - 1.0f;

    v8f acc0 = {};
    v8f acc1 = {};

    const int aBase = laneHi * 8;
    const int kb    = laneHi * 16;
    const int gOff  = kb + m;   // within-chunk K slot
    float*       slotW = smem + wave * 256 + (laneHi * 16 + m) * 8;
    const float* slotR = smem + wave * 256 + laneHi * 16 * 8;

    const int tile     = blockIdx.x;
    int       n        = counts[tile];
    n = n > CAP ? CAP : n;
    const int listBase = tile * CAP;

    // feature staging assignment: this thread fills Bls[nf][kk0..kk0+3]
    const int nf  = tid >> 3;
    const int kk0 = (tid & 7) * 4;
    const _Float16* fr = ft + (size_t)nf * NG;

    for (int c = 0; c < n; c += 32) {
        __syncthreads();                       // previous chunk fully consumed
        if (tid < 32) {
            int li = c + tid;
            idxLds[tid] = (li < n) ? list[listBase + li] : -1;
        }
        __syncthreads();                       // ids visible

        // ---- gather B tile: 32 gaussians x 32 features (shared by 8 waves)
        {
            int g0 = idxLds[kk0 + 0]; g0 = g0 < 0 ? 0 : g0;
            int g1 = idxLds[kk0 + 1]; g1 = g1 < 0 ? 0 : g1;
            int g2 = idxLds[kk0 + 2]; g2 = g2 < 0 ? 0 : g2;
            int g3 = idxLds[kk0 + 3]; g3 = g3 < 0 ? 0 : g3;
            _Float16* bw = Bls + nf * 32 + kk0;
            bw[0] = fr[g0];
            bw[1] = fr[g1];
            bw[2] = fr[g2];
            bw[3] = fr[g3];
        }

        // ---- phase 1: one listed gaussian per lane -> coefficients ----
        {
            int gi  = idxLds[gOff];
            int gld = gi < 0 ? 0 : gi;
            const float* p = gp + (size_t)gld * PSTR;
            v4f p0 = *(const v4f*)(p);         // mu0 mu1 mu2 lopac
            v4f p1 = *(const v4f*)(p + 4);     // A00 A11 A22 B01
            v4f p2 = *(const v4f*)(p + 8);     // B02 B12 s3x s3y
            float s3z = p[12];

            float d0 = px - p0.x;
            float d1 = py - p0.y;
            float c0 = (p1.x * d0 + p1.w * d1) * d0 + (p1.y * d1) * d1;
            float c1 = p2.x * d0 + p2.y * d1;
            bool ok = (gi >= 0) &
                      (__builtin_fabsf(d0) <= p2.z) &
                      (__builtin_fabsf(d1) <= p2.w);
            float e0 = ok ? (p0.w - c0) : -1.0e30f;

            v4f cc = { p0.z, p1.z, c1, e0 };   // mu2, A22, c1, e0
            *(v4f*)slotW = cc;
            slotW[4] = s3z;
        }
        __syncthreads();                       // B staged + slots written

        // ---- phase 2: build A tile ----
        v16h a;
#pragma unroll
        for (int e = 0; e < 16; ++e) {
            const float* sl = slotR + e * 8;   // same addr across group lanes
            v4f cc   = *(const v4f*)sl;        // mu2, A22, c1, e0
            float s3 = sl[4];
            float d2 = pz - cc.x;
            float tq = cc.y * d2 + cc.z;       // A22*d2 + c1
            float ea = cc.w - d2 * tq;         // lopac - q   (or -1e30)
            float w  = __builtin_exp2f(ea);
            a[e] = (__builtin_fabsf(d2) <= s3) ? (_Float16)w : (_Float16)0.0f;
        }

        v16h b0 = *(const v16h*)(Bls + m * 32 + kb);          // N = m
        v16h b1 = *(const v16h*)(Bls + (16 + m) * 32 + kb);   // N = 16+m

        acc0 = __builtin_amdgcn_wmma_f32_16x16x32_f16(
                   false, a, false, b0, (short)0, acc0, false, false);
        acc1 = __builtin_amdgcn_wmma_f32_16x16x32_f16(
                   false, a, false, b1, (short)0, acc1, false, false);
    }

    const int mOff = laneHi * 8;
#pragma unroll
    for (int r = 0; r < 8; ++r) {
        int M = r + mOff;
        float* orow = out + (size_t)(vBase + M) * NF;
        orow[m]      = acc0[r];
        orow[16 + m] = acc1[r];
    }
}

// ---------------------------------------------------------------------------
// Dense fallback (identical math, K loop over all 8192) if d_ws is too small
// for the binning buffers.
// ---------------------------------------------------------------------------
__global__ __launch_bounds__(256) void gv_main_dense(const float* __restrict__ gp,
                                                     const _Float16* __restrict__ ft,
                                                     float* __restrict__ out)
{
    __shared__ float smem[8 * 32 * 8];

    const int lane   = threadIdx.x & 31;
    const int wave   = threadIdx.x >> 5;
    const int laneHi = lane >> 4;
    const int m      = lane & 15;
    const int vBase  = blockIdx.x * 128 + wave * 16;
    const int v      = vBase + m;

    int i   = v / (WGRID * DGRID);
    int rem = v - i * (WGRID * DGRID);
    int j   = rem >> 4;
    int k   = rem & 15;
    float px = ((float)i + 0.5f) * 0.4f - 32.0f;
    float py = ((float)j + 0.5f) * 0.4f - 32.0f;
    float pz = ((float)k + 0.5f) * 0.4f - 1.0f;

    v8f acc0 = {};
    v8f acc1 = {};

    const int aBase = laneHi * 8;
    const int kb    = laneHi * 16;
    const _Float16* ftRow0 = ft + (size_t)m * NG + kb;
    const _Float16* ftRow1 = ft + (size_t)(16 + m) * NG + kb;

    const int gOff = kb + m;
    float*       slotW = smem + wave * 256 + (laneHi * 16 + m) * 8;
    const float* slotR = smem + wave * 256 + laneHi * 16 * 8;

    for (int chunk = 0; chunk < NG; chunk += 32) {
        {
            const float* p = gp + (size_t)(chunk + gOff) * PSTR;
            v4f p0 = *(const v4f*)(p);
            v4f p1 = *(const v4f*)(p + 4);
            v4f p2 = *(const v4f*)(p + 8);
            float s3z = p[12];

            float d0 = px - p0.x;
            float d1 = py - p0.y;
            float c0 = (p1.x * d0 + p1.w * d1) * d0 + (p1.y * d1) * d1;
            float c1 = p2.x * d0 + p2.y * d1;
            bool in01 = (__builtin_fabsf(d0) <= p2.z) &
                        (__builtin_fabsf(d1) <= p2.w);
            float e0 = in01 ? (p0.w - c0) : -1.0e30f;

            v4f cc = { p0.z, p1.z, c1, e0 };
            *(v4f*)slotW = cc;
            slotW[4] = s3z;
        }
        __builtin_amdgcn_wave_barrier();

        v16h a;
#pragma unroll
        for (int e = 0; e < 16; ++e) {
            const float* sl = slotR + e * 8;
            v4f cc   = *(const v4f*)sl;
            float s3 = sl[4];
            float d2 = pz - cc.x;
            float tq = cc.y * d2 + cc.z;
            float ea = cc.w - d2 * tq;
            float w  = __builtin_exp2f(ea);
            a[e] = (__builtin_fabsf(d2) <= s3) ? (_Float16)w : (_Float16)0.0f;
        }
        __builtin_amdgcn_wave_barrier();

        v16h b0 = *(const v16h*)(ftRow0 + chunk);
        v16h b1 = *(const v16h*)(ftRow1 + chunk);

        acc0 = __builtin_amdgcn_wmma_f32_16x16x32_f16(
                   false, a, false, b0, (short)0, acc0, false, false);
        acc1 = __builtin_amdgcn_wmma_f32_16x16x32_f16(
                   false, a, false, b1, (short)0, acc1, false, false);
    }

    const int mOff = laneHi * 8;
#pragma unroll
    for (int r = 0; r < 8; ++r) {
        int M = r + mOff;
        float* orow = out + (size_t)(vBase + M) * NF;
        orow[m]      = acc0[r];
        orow[16 + m] = acc1[r];
    }
}

// ---------------------------------------------------------------------------
extern "C" void kernel_launch(void* const* d_in, const int* in_sizes, int n_in,
                              void* d_out, int out_size, void* d_ws, size_t ws_size,
                              hipStream_t stream)
{
    const float* means  = (const float*)d_in[0];
    const float* opac   = (const float*)d_in[1];
    const float* scales = (const float*)d_in[2];
    const float* rot    = (const float*)d_in[3];
    const float* feat   = (const float*)d_in[4];

    char* ws = (char*)d_ws;
    float*    gp = (float*)ws;                              // 512 KB
    _Float16* ft = (_Float16*)(ws + NG * PSTR * sizeof(float));  // 512 KB
    size_t off   = NG * PSTR * sizeof(float) + NG * NF * sizeof(_Float16);
    int* counts  = (int*)(ws + off);                        // 12.8 KB (pad 16K)
    int* list    = (int*)(ws + off + 16384);                // 4 MB
    size_t need  = off + 16384 + (size_t)NTILE * CAP * sizeof(int);

    gv_prep<<<NG / 256, 256, 0, stream>>>(means, opac, scales, rot, gp);
    gv_featT<<<(NG * NF) / 256, 256, 0, stream>>>(feat, ft);

    if (ws_size >= need) {
        gv_zero<<<(NTILE + 255) / 256, 256, 0, stream>>>(counts);
        gv_count<<<NG / 256, 256, 0, stream>>>(gp, counts, list);
        gv_main_binned<<<NVOX / 128, 256, 0, stream>>>(gp, ft, counts, list,
                                                       (float*)d_out);
    } else {
        gv_main_dense<<<NVOX / 128, 256, 0, stream>>>(gp, ft, (float*)d_out);
    }
}
